// attention_64244120813878
// MI455X (gfx1250) — compile-verified, matrix operand order checked by reading, not run on hardware
//
#include <hip/hip_runtime.h>

// B=4, S=2048, D=768, H=12, HD=64, SCALE = 64^-0.5 = 0.125
#define BB 4
#define SS 2048
#define DD 768
#define HH 12
#define HDIM 64
#define SCALE 0.125f

typedef __attribute__((ext_vector_type(16))) __bf16 v16bf;
typedef __attribute__((ext_vector_type(8)))  float  v8f;

union H8 {          // 16 bytes = 8 bf16
  uint4  q;
  __bf16 bf[8];
};

// ---------------------------------------------------------------------------
// Fragment loaders per CDNA5 ISA 7.12.2 (16-bit A/B layout, wave32):
//   element (row, k): lane = row + 16*((k/8)&1), slot = (k%8) + 8*(k/16)
// => per lane: slots 0-7  = src[row][8*h .. 8*h+7]
//              slots 8-15 = src[row][16+8*h .. 16+8*h+7],  h = lane>>4
// ---------------------------------------------------------------------------
__device__ __forceinline__ v16bf frag_b16(const __bf16* base, int stride, int lane) {
  const int r  = lane & 15;
  const int h8 = (lane >> 4) << 3;
  const __bf16* p = base + r * stride + h8;
  H8 lo, hi;
  lo.q = *(const uint4*)(p);
  hi.q = *(const uint4*)(p + 16);
  v16bf f;
#pragma unroll
  for (int i = 0; i < 8; ++i) { f[i] = lo.bf[i]; f[i + 8] = hi.bf[i]; }
  return f;
}

__device__ __forceinline__ v16bf frag_f32cvt(const float* base, int stride, int lane) {
  const int r  = lane & 15;
  const int h8 = (lane >> 4) << 3;
  const float* p = base + r * stride + h8;
  const float4 a0 = *(const float4*)(p);
  const float4 a1 = *(const float4*)(p + 4);
  const float4 b0 = *(const float4*)(p + 16);
  const float4 b1 = *(const float4*)(p + 20);
  v16bf f;
  f[0]  = (__bf16)a0.x; f[1]  = (__bf16)a0.y; f[2]  = (__bf16)a0.z; f[3]  = (__bf16)a0.w;
  f[4]  = (__bf16)a1.x; f[5]  = (__bf16)a1.y; f[6]  = (__bf16)a1.z; f[7]  = (__bf16)a1.w;
  f[8]  = (__bf16)b0.x; f[9]  = (__bf16)b0.y; f[10] = (__bf16)b0.z; f[11] = (__bf16)b0.w;
  f[12] = (__bf16)b1.x; f[13] = (__bf16)b1.y; f[14] = (__bf16)b1.z; f[15] = (__bf16)b1.w;
  return f;
}

__device__ __forceinline__ v8f wmma_bf16(v16bf a, v16bf b, v8f c) {
  return __builtin_amdgcn_wmma_f32_16x16x32_bf16(false, a, false, b, (short)0, c,
                                                 false, false);
}

// CDNA5 async global->LDS copy (ASYNCcnt-tracked), ISA 15.18.3 op 98.
__device__ __forceinline__ void async_copy_b128(unsigned lds_off, const void* gptr) {
  asm volatile("global_load_async_to_lds_b128 %0, %1, off"
               :: "v"(lds_off), "v"(gptr) : "memory");
}
__device__ __forceinline__ void wait_async0() {
  asm volatile("s_wait_asynccnt 0x0" ::: "memory");
}

// ---------------------------------------------------------------------------
// QKV projection: out[b][head][s][hd] = bf16( (x@W + bias)[row][head*64+hd]*scale )
// grid (32, 12), block 256 (8 waves). Tile M=256 (32 rows/wave), N=64, K-step 32.
// ---------------------------------------------------------------------------
__global__ __launch_bounds__(256) void qkv_gemm(const float* __restrict__ X,
                                                const float* __restrict__ W,
                                                const float* __restrict__ bias,
                                                __bf16* __restrict__ out,
                                                float scale) {
  __shared__ __bf16 sBT[64][40];   // W tile transposed: [n][k], padded stride
  const int tid  = threadIdx.x;
  const int wave = tid >> 5, lane = tid & 31;
  const int lr = lane & 15, lh = lane >> 4;
  const int row0 = blockIdx.x * 256 + wave * 32;
  const int head = blockIdx.y;

  v8f acc[2][4] = {};

  for (int kk = 0; kk < DD; kk += 32) {
    // stage 32x64 W tile transposed (2048 f32 = 512 float4, 2 per thread)
    for (int i = tid; i < 512; i += 256) {
      const int r  = i >> 4;            // k-local 0..31
      const int c4 = (i & 15) << 2;     // n-local 0..60
      const float* wp = W + (size_t)(kk + r) * DD + head * 64 + c4;
      if (kk + 32 < DD) __builtin_prefetch(wp + (size_t)32 * DD, 0, 0);
      const float4 w4 = *(const float4*)(wp);
      sBT[c4 + 0][r] = (__bf16)w4.x;
      sBT[c4 + 1][r] = (__bf16)w4.y;
      sBT[c4 + 2][r] = (__bf16)w4.z;
      sBT[c4 + 3][r] = (__bf16)w4.w;
    }
    __syncthreads();
    const v16bf af0 = frag_f32cvt(X + (size_t)row0 * DD + kk, DD, lane);
    const v16bf af1 = frag_f32cvt(X + (size_t)(row0 + 16) * DD + kk, DD, lane);
    v16bf bf[4];
#pragma unroll
    for (int nt = 0; nt < 4; ++nt) bf[nt] = frag_b16(&sBT[nt * 16][0], 40, lane);
#pragma unroll
    for (int nt = 0; nt < 4; ++nt) {
      acc[0][nt] = wmma_bf16(af0, bf[nt], acc[0][nt]);
      acc[1][nt] = wmma_bf16(af1, bf[nt], acc[1][nt]);
    }
    __syncthreads();
  }

  // epilogue: C layout lane = n + 16*(m/8), vgpr = m%8
#pragma unroll
  for (int nt = 0; nt < 4; ++nt) {
    const int hd = nt * 16 + lr;
    const float bv = bias[head * 64 + hd];
#pragma unroll
    for (int g = 0; g < 2; ++g)
#pragma unroll
      for (int v = 0; v < 8; ++v) {
        const int row = row0 + g * 16 + v + 8 * lh;
        const int b = row >> 11, s = row & (SS - 1);
        out[(((size_t)b * HH + head) * SS + s) * HDIM + hd] =
            (__bf16)((acc[g][nt][v] + bv) * scale);
      }
  }
}

// ---------------------------------------------------------------------------
// Flash attention: grid (32, 12, 4), block 128 (4 waves x 16 Q rows = 64 rows).
// Key chunks of 64; K staged via async-to-LDS, V^T via manual transpose;
// online softmax; P relayout via per-wave LDS scratch.
// Output bf16 [B, S, 768] (heads merged) feeding the final projection.
// ---------------------------------------------------------------------------
__global__ __launch_bounds__(128) void flash_attn(const __bf16* __restrict__ Q,
                                                  const __bf16* __restrict__ K,
                                                  const __bf16* __restrict__ V,
                                                  const float* __restrict__ mask,
                                                  __bf16* __restrict__ O) {
  __shared__ __bf16 sK[64][72];       // [key][feat]
  __shared__ __bf16 sVT[64][72];      // [feat][key]
  __shared__ __bf16 sP[4][16][72];    // per-wave P scratch [m][key]

  const int tid  = threadIdx.x;
  const int wave = tid >> 5, lane = tid & 31;
  const int lr = lane & 15, lh = lane >> 4;
  const int h = blockIdx.y, b = blockIdx.z;
  const size_t bh = (size_t)b * HH + h;

  const __bf16* Qb = Q + (bh * SS + blockIdx.x * 64 + wave * 16) * HDIM;
  const __bf16* Kb = K + bh * SS * HDIM;
  const __bf16* Vb = V + bh * SS * HDIM;
  const float*  mb = mask + (size_t)b * SS;

  const v16bf qf0 = frag_b16(Qb, HDIM, lane);        // features 0..31
  const v16bf qf1 = frag_b16(Qb + 32, HDIM, lane);   // features 32..63

  v8f o[4] = {};
  float mrow[8], lrow[8];
#pragma unroll
  for (int v = 0; v < 8; ++v) { mrow[v] = -3.0e38f; lrow[v] = 0.f; }

  for (int key0 = 0; key0 < SS; key0 += 64) {
    // stage K (async DMA) and V^T (manual transpose): 64x64 bf16 each
    for (int i = tid; i < 512; i += 128) {
      const int r  = i >> 3;          // key-local 0..63
      const int c8 = (i & 7) << 3;    // feat-local 0..56
      async_copy_b128((unsigned)(size_t)&sK[r][c8],
                      Kb + (size_t)(key0 + r) * HDIM + c8);
      H8 vv; vv.q = *(const uint4*)(Vb + (size_t)(key0 + r) * HDIM + c8);
#pragma unroll
      for (int j = 0; j < 8; ++j) sVT[c8 + j][r] = vv.bf[j];
    }
    if (key0 + 64 < SS) {             // warm L2/L0 for next chunk
      __builtin_prefetch(Kb + (size_t)(key0 + 64) * HDIM + tid * 32, 0, 0);
      __builtin_prefetch(Vb + (size_t)(key0 + 64) * HDIM + tid * 32, 0, 0);
    }
    wait_async0();
    __syncthreads();

    // scores: 16 q-rows x 64 keys (four 16x16 C tiles), K-dim = 64 features
    v8f c[4] = {};
#pragma unroll
    for (int nt = 0; nt < 4; ++nt) {
      const v16bf k0 = frag_b16(&sK[nt * 16][0],      72, lane);
      const v16bf k1 = frag_b16(&sK[nt * 16][0] + 32, 72, lane);
      c[nt] = wmma_bf16(qf0, k0, c[nt]);
      c[nt] = wmma_bf16(qf1, k1, c[nt]);
    }

    float mk[4];
#pragma unroll
    for (int t = 0; t < 4; ++t) mk[t] = mb[key0 + t * 16 + lr];

#pragma unroll
    for (int v = 0; v < 8; ++v) {
      float s[4];
#pragma unroll
      for (int t = 0; t < 4; ++t) s[t] = c[t][v] + mk[t];
      float mc = fmaxf(fmaxf(s[0], s[1]), fmaxf(s[2], s[3]));
      mc = fmaxf(mc, __shfl_xor(mc, 1, 32));
      mc = fmaxf(mc, __shfl_xor(mc, 2, 32));
      mc = fmaxf(mc, __shfl_xor(mc, 4, 32));
      mc = fmaxf(mc, __shfl_xor(mc, 8, 32));
      const float mnew  = fmaxf(mrow[v], mc);
      const float alpha = __expf(mrow[v] - mnew);
      float p[4], ls = 0.f;
#pragma unroll
      for (int t = 0; t < 4; ++t) { p[t] = __expf(s[t] - mnew); ls += p[t]; }
      ls += __shfl_xor(ls, 1, 32);
      ls += __shfl_xor(ls, 2, 32);
      ls += __shfl_xor(ls, 4, 32);
      ls += __shfl_xor(ls, 8, 32);
      lrow[v] = lrow[v] * alpha + ls;
      mrow[v] = mnew;
#pragma unroll
      for (int nt = 0; nt < 4; ++nt) o[nt][v] *= alpha;
      const int m = v + 8 * lh;                       // C-layout row of this lane
#pragma unroll
      for (int t = 0; t < 4; ++t) sP[wave][m][lr + t * 16] = (__bf16)p[t];
    }
    asm volatile("" ::: "memory");   // keep sP stores before the gathers below
                                     // (LDS ops of one wave execute in order)

    // P(16x64) @ V(64x64)
    const v16bf pf0 = frag_b16(&sP[wave][0][0],      72, lane);
    const v16bf pf1 = frag_b16(&sP[wave][0][0] + 32, 72, lane);
#pragma unroll
    for (int nt = 0; nt < 4; ++nt) {
      const v16bf vf0 = frag_b16(&sVT[nt * 16][0],      72, lane);
      const v16bf vf1 = frag_b16(&sVT[nt * 16][0] + 32, 72, lane);
      o[nt] = wmma_bf16(pf0, vf0, o[nt]);
      o[nt] = wmma_bf16(pf1, vf1, o[nt]);
    }
    __syncthreads();
  }

  // normalize + store merged-head bf16 [B, S, 768]
  const int s_base = blockIdx.x * 64 + wave * 16;
#pragma unroll
  for (int v = 0; v < 8; ++v) {
    const float inv = 1.f / lrow[v];
    const int s = s_base + v + 8 * lh;
#pragma unroll
    for (int nt = 0; nt < 4; ++nt) {
      const int col = h * 64 + nt * 16 + lr;
      O[((size_t)b * SS + s) * DD + col] = (__bf16)(o[nt][v] * inv);
    }
  }
}

// ---------------------------------------------------------------------------
// Output projection: out(f32) = attn(bf16) @ Wo + bo. grid (32, 12), block 256.
// ---------------------------------------------------------------------------
__global__ __launch_bounds__(256) void out_proj(const __bf16* __restrict__ A,
                                                const float* __restrict__ W,
                                                const float* __restrict__ bias,
                                                float* __restrict__ out) {
  __shared__ __bf16 sBT[64][40];
  const int tid  = threadIdx.x;
  const int wave = tid >> 5, lane = tid & 31;
  const int lr = lane & 15, lh = lane >> 4;
  const int row0 = blockIdx.x * 256 + wave * 32;
  const int nt0  = blockIdx.y;

  v8f acc[2][4] = {};

  for (int kk = 0; kk < DD; kk += 32) {
    for (int i = tid; i < 512; i += 256) {
      const int r  = i >> 4;
      const int c4 = (i & 15) << 2;
      const float* wp = W + (size_t)(kk + r) * DD + nt0 * 64 + c4;
      if (kk + 32 < DD) __builtin_prefetch(wp + (size_t)32 * DD, 0, 0);
      const float4 w4 = *(const float4*)(wp);
      sBT[c4 + 0][r] = (__bf16)w4.x;
      sBT[c4 + 1][r] = (__bf16)w4.y;
      sBT[c4 + 2][r] = (__bf16)w4.z;
      sBT[c4 + 3][r] = (__bf16)w4.w;
    }
    __syncthreads();
    const v16bf af0 = frag_b16(A + (size_t)row0 * DD + kk, DD, lane);
    const v16bf af1 = frag_b16(A + (size_t)(row0 + 16) * DD + kk, DD, lane);
    v16bf bf[4];
#pragma unroll
    for (int nt = 0; nt < 4; ++nt) bf[nt] = frag_b16(&sBT[nt * 16][0], 40, lane);
#pragma unroll
    for (int nt = 0; nt < 4; ++nt) {
      acc[0][nt] = wmma_bf16(af0, bf[nt], acc[0][nt]);
      acc[1][nt] = wmma_bf16(af1, bf[nt], acc[1][nt]);
    }
    __syncthreads();
  }

#pragma unroll
  for (int nt = 0; nt < 4; ++nt) {
    const int col = nt0 * 64 + nt * 16 + lr;
    const float bv = bias[col];
#pragma unroll
    for (int g = 0; g < 2; ++g)
#pragma unroll
      for (int v = 0; v < 8; ++v) {
        const int row = row0 + g * 16 + v + 8 * lh;
        out[(size_t)row * DD + col] = acc[g][nt][v] + bv;
      }
  }
}

// ---------------------------------------------------------------------------
extern "C" void kernel_launch(void* const* d_in, const int* in_sizes, int n_in,
                              void* d_out, int out_size, void* d_ws, size_t ws_size,
                              hipStream_t stream) {
  (void)in_sizes; (void)n_in; (void)out_size; (void)ws_size;
  const float* x    = (const float*)d_in[0];
  const float* am   = (const float*)d_in[1];
  const float* Wq   = (const float*)d_in[2];
  const float* bq   = (const float*)d_in[3];
  const float* Wk   = (const float*)d_in[4];
  const float* bk   = (const float*)d_in[5];
  const float* Wv   = (const float*)d_in[6];
  const float* bv   = (const float*)d_in[7];
  const float* Wo   = (const float*)d_in[8];
  const float* bo   = (const float*)d_in[9];
  float* out = (float*)d_out;

  const size_t nqkv = (size_t)BB * HH * SS * HDIM;   // 6,291,456 elems
  __bf16* q    = (__bf16*)d_ws;
  __bf16* k    = q + nqkv;
  __bf16* v    = k + nqkv;
  __bf16* attn = v + nqkv;                           // [B, S, 768] bf16

  const dim3 gemm_grid(32, 12);
  qkv_gemm<<<gemm_grid, 256, 0, stream>>>(x, Wq, bq, q, SCALE);
  qkv_gemm<<<gemm_grid, 256, 0, stream>>>(x, Wk, bk, k, 1.0f);
  qkv_gemm<<<gemm_grid, 256, 0, stream>>>(x, Wv, bv, v, 1.0f);

  flash_attn<<<dim3(32, 12, 4), 128, 0, stream>>>(q, k, v, am, attn);

  out_proj<<<gemm_grid, 256, 0, stream>>>(attn, Wo, bo, out);
}
